// SAGE_29686813950021
// MI455X (gfx1250) — matrix operand order repaired and checked by reading, not compiled, not run on hardware
//
#include <hip/hip_runtime.h>
#include <hip/hip_bf16.h>

// ---------------------------------------------------------------------------
// 3-layer GraphSAGE inference for MI455X (gfx1250, wave32).
//  - scatter-mean aggregation: f32 global atomics (memory/atomic bound part)
//  - dense lin_l/lin_r GEMMs:  v_wmma_f32_16x16x32_bf16, f32 accumulate
//  - BN+ReLU fused into GEMM epilogue
// ---------------------------------------------------------------------------

typedef __attribute__((ext_vector_type(16))) __bf16 v16bf;
typedef __attribute__((ext_vector_type(8)))  __bf16 v8bf;
typedef __attribute__((ext_vector_type(4)))  __bf16 v4bf;
typedef __attribute__((ext_vector_type(8)))  float  v8f;

#define KDIM 128          // input feature dim of every layer's GEMM (DIN == H == 128)

// ---------------------------------------------------------------------------
// utility: zero a float buffer (graph-capture-safe replacement for memset)
// ---------------------------------------------------------------------------
__global__ void zero_f32(float* __restrict__ p, size_t n) {
    size_t i = (size_t)blockIdx.x * blockDim.x + threadIdx.x;
    size_t stride = (size_t)gridDim.x * blockDim.x;
    for (; i < n; i += stride) p[i] = 0.0f;
}

// ---------------------------------------------------------------------------
// degree histogram + inverse (shared by all 3 layers)
// ---------------------------------------------------------------------------
__global__ void degree_count(const int* __restrict__ dst, float* __restrict__ deg, int E) {
    int e = blockIdx.x * blockDim.x + threadIdx.x;
    if (e < E) atomicAdd(&deg[dst[e]], 1.0f);
}

__global__ void finalize_invdeg(float* __restrict__ deg, int n) {
    int i = blockIdx.x * blockDim.x + threadIdx.x;
    if (i < n) deg[i] = 1.0f / fmaxf(deg[i], 1.0f);
}

// ---------------------------------------------------------------------------
// Repack a f32 weight matrix W[K=128][nout] into per-lane WMMA B fragments
// (bf16).  Fragment f = ct*4 + kt covers columns [ct*16, ct*16+16) and
// K in [kt*32, kt*32+32).  Per ISA 7.12.2, lane L of a 16-bit 32x16 B
// fragment holds column N = L&15, K = (L>>4)*8 + (e&7) + (e>>3)*16  (e = 0..15).
// Layout in memory: packed[f*512 + lane*16 + e]  ->  each lane's fragment is
// a contiguous 32-byte run -> two global_load_b128 in the GEMM.
// ---------------------------------------------------------------------------
__global__ void pack_weight_bf16(const float* __restrict__ W, __bf16* __restrict__ out, int nout) {
    int tid = blockIdx.x * blockDim.x + threadIdx.x;
    int total = KDIM * nout;
    if (tid >= total) return;
    int e    = tid & 15;
    int lane = (tid >> 4) & 31;
    int f    = tid >> 9;
    int kt   = f & 3;       // K tile (K = 128 -> 4 steps of 32)
    int ct   = f >> 2;      // column tile
    int n    = ct * 16 + (lane & 15);
    int k    = kt * 32 + (lane >> 4) * 8 + (e & 7) + (e >> 3) * 16;
    out[tid] = (__bf16)W[k * nout + n];
}

// ---------------------------------------------------------------------------
// Edge scatter: agg[dst] += x[src]  (f32 atomics).  32 threads per edge,
// each handling 4 consecutive features via a float4 gather.
// ---------------------------------------------------------------------------
__global__ void scatter_mean_accum(const float* __restrict__ x,
                                   const int* __restrict__ src,
                                   const int* __restrict__ dst,
                                   float* __restrict__ agg, int E) {
    long long tid = (long long)blockIdx.x * blockDim.x + threadIdx.x;
    int e = (int)(tid >> 5);
    if (e >= E) return;
    int c = ((int)tid & 31) << 2;
    int s = src[e], d = dst[e];
    const float4 v = *(const float4*)(x + (size_t)s * KDIM + c);
    float* o = agg + (size_t)d * KDIM + c;
    atomicAdd(o + 0, v.x);
    atomicAdd(o + 1, v.y);
    atomicAdd(o + 2, v.z);
    atomicAdd(o + 3, v.w);
}

// ---------------------------------------------------------------------------
// A-fragment load from LDS, per ISA 7.12.2 16-bit A 16x32 layout:
// lane L: row M = L&15; half-wave selects K base; the lane's 16 halves are
// two contiguous runs of 8:  K = kb..kb+7 and K = kb+16..kb+23  (kb = hs*8).
// With a 16-byte row pad, both runs are 16B-aligned -> 2x ds_load_b128.
// ---------------------------------------------------------------------------
__device__ inline v16bf load_fragA(const __bf16* p) {
    v8bf lo = *(const v8bf*)(p);
    v8bf hi = *(const v8bf*)(p + 16);
    v16bf r;
#pragma unroll
    for (int e = 0; e < 8; ++e) { r[e] = lo[e]; r[e + 8] = hi[e]; }
    return r;
}

// ---------------------------------------------------------------------------
// Fused SAGE layer GEMM:
//   out = bn_relu( (agg * invdeg) @ Wl + bl + x @ Wr )
// 128 threads = 4 waves; block covers 64 rows; each wave owns a 16-row stripe
// and loops over 16-column output tiles.  K = 128 -> 4 WMMA k-steps, two
// matrices -> 8 v_wmma_f32_16x16x32_bf16 per output tile.
// ---------------------------------------------------------------------------
template <int NOUT, bool BNRELU>
__global__ __launch_bounds__(128)
void sage_layer_gemm(const float* __restrict__ agg,
                     const float* __restrict__ invdeg,
                     const float* __restrict__ x,
                     const __bf16* __restrict__ Wl,   // packed fragments
                     const __bf16* __restrict__ Wr,   // packed fragments
                     const float* __restrict__ bl,
                     const float* __restrict__ g,  const float* __restrict__ be,
                     const float* __restrict__ rm, const float* __restrict__ rv,
                     float* __restrict__ out, int n) {
    constexpr int LDA = KDIM + 8;                 // +16B pad: conflict-free b128 reads
    __shared__ alignas(16) __bf16 sA[64 * LDA];   // invdeg-scaled aggregate tile
    __shared__ alignas(16) __bf16 sX[64 * LDA];   // root feature tile

    const int rowBase = blockIdx.x * 64;
    const int t = threadIdx.x;

    // ---- cooperative stage: 64 rows x 128 cols, f32 -> bf16 (agg scaled) ----
    for (int i = t; i < 64 * (KDIM / 4); i += 128) {
        int r = i >> 5;                 // row in tile
        int c = (i & 31) << 2;          // feature (float4 granularity)
        int gr = rowBase + r;
        float4 va = make_float4(0.f, 0.f, 0.f, 0.f);
        float4 vx = make_float4(0.f, 0.f, 0.f, 0.f);
        float sc = 0.f;
        if (gr < n) {
            va = *(const float4*)(agg + (size_t)gr * KDIM + c);
            vx = *(const float4*)(x   + (size_t)gr * KDIM + c);
            sc = invdeg[gr];
        }
        v4bf pa = { (__bf16)(va.x * sc), (__bf16)(va.y * sc),
                    (__bf16)(va.z * sc), (__bf16)(va.w * sc) };
        v4bf px = { (__bf16)vx.x, (__bf16)vx.y, (__bf16)vx.z, (__bf16)vx.w };
        *(v4bf*)&sA[r * LDA + c] = pa;   // 8-byte ds_store
        *(v4bf*)&sX[r * LDA + c] = px;
    }
    __syncthreads();

    const int wave = t >> 5;
    const int lane = t & 31;
    const int m  = lane & 15;            // matrix row (A) / column (B,C)
    const int hs = lane >> 4;            // half-wave select
    const int ldsRow = wave * 16 + m;

#pragma unroll
    for (int ct = 0; ct < NOUT / 16; ++ct) {
        v8f acc = {};
#pragma unroll
        for (int kt = 0; kt < 4; ++kt) {
            const int k0 = kt * 32 + hs * 8;
            const int fbase = (ct * 4 + kt) * 512 + lane * 16;
            // lin_l on scaled aggregate
            v16bf a0 = load_fragA(&sA[ldsRow * LDA + k0]);
            v16bf b0 = *(const v16bf*)(Wl + fbase);
            acc = __builtin_amdgcn_wmma_f32_16x16x32_bf16(
                      false, a0, false, b0, (short)0, acc, false, false);
            // lin_r on root features
            v16bf a1 = load_fragA(&sX[ldsRow * LDA + k0]);
            v16bf b1 = *(const v16bf*)(Wr + fbase);
            acc = __builtin_amdgcn_wmma_f32_16x16x32_bf16(
                      false, a1, false, b1, (short)0, acc, false, false);
        }
        // ---- epilogue: + bl, optional BN(eval) + ReLU, store f32 ----
        const int col = ct * 16 + m;
        const float bias = bl[col];
        float scale = 1.f, shift = 0.f;
        if (BNRELU) {
            scale = g[col] * rsqrtf(rv[col] + 1e-5f);
            shift = be[col] - rm[col] * scale;
        }
#pragma unroll
        for (int v = 0; v < 8; ++v) {
            int gr = rowBase + wave * 16 + v + 8 * hs;   // C/D layout: M = v + 8*hs
            if (gr < n) {
                float val = acc[v] + bias;
                if (BNRELU) val = fmaxf(val * scale + shift, 0.f);
                out[(size_t)gr * NOUT + col] = val;
            }
        }
    }
}

// ---------------------------------------------------------------------------
// Host-side orchestration (graph-capture safe: kernels only, all on stream)
// ---------------------------------------------------------------------------
extern "C" void kernel_launch(void* const* d_in, const int* in_sizes, int n_in,
                              void* d_out, int out_size, void* d_ws, size_t ws_size,
                              hipStream_t stream) {
    const int N = in_sizes[0] / KDIM;   // 100000
    const int E = in_sizes[1] / 2;      // 600000

    const float* x   = (const float*)d_in[0];
    const int*   src = (const int*)d_in[1];
    const int*   dst = src + E;
    const float* Wl1 = (const float*)d_in[2];  const float* bl1 = (const float*)d_in[3];
    const float* Wr1 = (const float*)d_in[4];
    const float* g1  = (const float*)d_in[5];  const float* be1 = (const float*)d_in[6];
    const float* rm1 = (const float*)d_in[7];  const float* rv1 = (const float*)d_in[8];
    const float* Wl2 = (const float*)d_in[9];  const float* bl2 = (const float*)d_in[10];
    const float* Wr2 = (const float*)d_in[11];
    const float* g2  = (const float*)d_in[12]; const float* be2 = (const float*)d_in[13];
    const float* rm2 = (const float*)d_in[14]; const float* rv2 = (const float*)d_in[15];
    const float* Wl3 = (const float*)d_in[16]; const float* bl3 = (const float*)d_in[17];
    const float* Wr3 = (const float*)d_in[18];

    // -------- workspace carve-out (~154 MB) --------
    char* ws = (char*)d_ws;
    auto carve = [&](size_t bytes) -> char* {
        char* p = ws;
        ws += (bytes + 255) & ~(size_t)255;
        return p;
    };
    float*  inv  = (float*)carve((size_t)N * sizeof(float));
    float*  agg  = (float*)carve((size_t)N * KDIM * sizeof(float));
    float*  h1   = (float*)carve((size_t)N * KDIM * sizeof(float));
    float*  h2   = (float*)carve((size_t)N * KDIM * sizeof(float));
    __bf16* pWl1 = (__bf16*)carve((size_t)KDIM * 128 * 2);
    __bf16* pWr1 = (__bf16*)carve((size_t)KDIM * 128 * 2);
    __bf16* pWl2 = (__bf16*)carve((size_t)KDIM * 128 * 2);
    __bf16* pWr2 = (__bf16*)carve((size_t)KDIM * 128 * 2);
    __bf16* pWl3 = (__bf16*)carve((size_t)KDIM * 64 * 2);
    __bf16* pWr3 = (__bf16*)carve((size_t)KDIM * 64 * 2);

    // -------- degrees (shared across layers) --------
    zero_f32<<<1024, 256, 0, stream>>>(inv, (size_t)N);
    degree_count<<<(E + 255) / 256, 256, 0, stream>>>(dst, inv, E);
    finalize_invdeg<<<(N + 255) / 256, 256, 0, stream>>>(inv, N);

    // -------- one-time bf16 weight repack into WMMA fragment order --------
    pack_weight_bf16<<<(KDIM * 128 + 255) / 256, 256, 0, stream>>>(Wl1, pWl1, 128);
    pack_weight_bf16<<<(KDIM * 128 + 255) / 256, 256, 0, stream>>>(Wr1, pWr1, 128);
    pack_weight_bf16<<<(KDIM * 128 + 255) / 256, 256, 0, stream>>>(Wl2, pWl2, 128);
    pack_weight_bf16<<<(KDIM * 128 + 255) / 256, 256, 0, stream>>>(Wr2, pWr2, 128);
    pack_weight_bf16<<<(KDIM * 64 + 255) / 256, 256, 0, stream>>>(Wl3, pWl3, 64);
    pack_weight_bf16<<<(KDIM * 64 + 255) / 256, 256, 0, stream>>>(Wr3, pWr3, 64);

    const int scatterBlocks = (int)(((long long)E * 32 + 255) / 256);
    const int gemmBlocks = (N + 63) / 64;

    // -------- layer 1 --------
    zero_f32<<<4096, 256, 0, stream>>>(agg, (size_t)N * KDIM);
    scatter_mean_accum<<<scatterBlocks, 256, 0, stream>>>(x, src, dst, agg, E);
    sage_layer_gemm<128, true><<<gemmBlocks, 128, 0, stream>>>(
        agg, inv, x, pWl1, pWr1, bl1, g1, be1, rm1, rv1, h1, N);

    // -------- layer 2 --------
    zero_f32<<<4096, 256, 0, stream>>>(agg, (size_t)N * KDIM);
    scatter_mean_accum<<<scatterBlocks, 256, 0, stream>>>(h1, src, dst, agg, E);
    sage_layer_gemm<128, true><<<gemmBlocks, 128, 0, stream>>>(
        agg, inv, h1, pWl2, pWr2, bl2, g2, be2, rm2, rv2, h2, N);

    // -------- layer 3 (no BN/ReLU) --------
    zero_f32<<<4096, 256, 0, stream>>>(agg, (size_t)N * KDIM);
    scatter_mean_accum<<<scatterBlocks, 256, 0, stream>>>(h2, src, dst, agg, E);
    sage_layer_gemm<64, false><<<gemmBlocks, 128, 0, stream>>>(
        agg, inv, h2, pWl3, pWr3, bl3, nullptr, nullptr, nullptr, nullptr,
        (float*)d_out, N);
}
